// MultiheadSelfAttention_82300163326105
// MI455X (gfx1250) — matrix operand order repaired and checked by reading, not compiled, not run on hardware
//
#include <hip/hip_runtime.h>
#include <hip/hip_bf16.h>
#include <math.h>

#define D_MODEL 1024
#define NUM_HEADS 16
#define D_K 64
#define BATCH 2
#define SEQ 2048
#define THETA 10000.0f
#define NEG_BIG (-3.0e30f)

typedef __bf16 bf16_t;
typedef __attribute__((ext_vector_type(8)))  __bf16 v8bf;
typedef __attribute__((ext_vector_type(16))) __bf16 v16bf;
typedef __attribute__((ext_vector_type(8)))  float  v8f;
typedef __attribute__((ext_vector_type(4)))  unsigned int v4u;
typedef __attribute__((ext_vector_type(8)))  int v8i;
typedef __attribute__((ext_vector_type(4)))  int v4i;

static __device__ __forceinline__ v16bf frag_cat(v8bf lo, v8bf hi) {
  return __builtin_shufflevector(lo, hi, 0,1,2,3,4,5,6,7,8,9,10,11,12,13,14,15);
}

static __device__ __forceinline__ v8f wmma_bf16(v16bf a, v16bf b, v8f c) {
  // D = A(16x32 bf16) * B(32x16 bf16) + C(16x16 f32)
  return __builtin_amdgcn_wmma_f32_16x16x32_bf16(false, a, false, b, (short)0, c,
                                                 false, false);
}

// ---------------------------------------------------------------------------
// TDM: stage a [128 rows x 32 k] bf16 tile (row stride = K elements) into LDS.
// Descriptor per CDNA5 ISA section 8.3/8.4 (2-D tile, groups 2/3 zero).
// 6-arg toolchain form: (g0, g1, g2, g3, g_extra, cpol).
// ---------------------------------------------------------------------------
static __device__ __forceinline__ void tdm_load_tile_128x32(unsigned lds_off,
                                                            unsigned long long gaddr) {
  v4u g0 = { 1u,                                   // count=1 valid descriptor
             lds_off,                              // lds_addr (bytes)
             (unsigned)(gaddr & 0xffffffffu),      // global_addr[31:0]
             (unsigned)((gaddr >> 32) & 0x01ffffffu) | 0x80000000u };  // +type=2
  v8i g1 = { (int)0x00010000,        // wg_mask=0, data_size=1 (2 bytes)
             (int)(1024u << 16),     // tensor_dim0[15:0]=1024 (atomic_addr=0)
             (int)(1024u << 16),     // tensor_dim0 hi=0 | tensor_dim1[15:0]=1024
             (int)(32u << 16),       // tensor_dim1 hi=0 | tile_dim0=32
             (int)128,               // tile_dim1=128, tile_dim2=0
             (int)1024,              // tensor_dim0_stride = K elements
             0, 0 };                 // stride hi / dim1_stride (unused for 2D)
  v4i gz4 = {0, 0, 0, 0};
  v8i gz8 = {0, 0, 0, 0, 0, 0, 0, 0};
  __builtin_amdgcn_tensor_load_to_lds(g0, g1, gz4, gz4, gz8, 0);
}

// ---------------------------------------------------------------------------
// fp32 -> bf16 conversion
// ---------------------------------------------------------------------------
__global__ void f32_to_bf16_kernel(const float* __restrict__ src,
                                   bf16_t* __restrict__ dst, int n) {
  int i = blockIdx.x * blockDim.x + threadIdx.x;
  if (i < n) dst[i] = (bf16_t)src[i];
}

// ---------------------------------------------------------------------------
// C[m,n] = sum_k A[m,k] * W[n,k]   (x @ W.T style), TDM-staged weights.
// Block = 128 threads (4 waves) computes a 64(m) x 128(n) tile.
// Wave w: rows m0 = blk_m + (w&1)*32 (two 16-row subtiles),
//         cols n  = blk_n + (w>>1)*64 (four 16-col fragments).
// W k-chunk tile [128 rows x 32 k] is DMA'd into LDS by wave 0 via
// tensor_load_to_lds, double buffered, TENSORcnt-synchronized.
// ---------------------------------------------------------------------------
__global__ void gemm_xwt_kernel(const bf16_t* __restrict__ A,
                                const bf16_t* __restrict__ W,
                                float* __restrict__ C, int M, int N, int K) {
  __shared__ __attribute__((aligned(16))) bf16_t Wtile[2][128][32];

  const int lane = threadIdx.x & 31;
  const int wv   = threadIdx.x >> 5;
  const int tiles_n = N >> 7;
  const int bm = (int)blockIdx.x / tiles_n;
  const int bn = (int)blockIdx.x % tiles_n;
  const int m0 = bm * 64 + (wv & 1) * 32;
  const int n0blk = bn * 128;
  const int nw = (wv >> 1) * 64;          // wave's col offset inside LDS tile
  const int half = lane >> 4;
  const int lr   = lane & 15;

  const bf16_t* arow0 = A + (size_t)(m0 + lr) * K;
  const bf16_t* arow1 = A + (size_t)(m0 + 16 + lr) * K;
  const unsigned long long wbase =
      (unsigned long long)(uintptr_t)(W + (size_t)n0blk * K);
  const unsigned lds0 = (unsigned)(uintptr_t)(&Wtile[0][0][0]);
  const unsigned lds1 = (unsigned)(uintptr_t)(&Wtile[1][0][0]);

  const v8f zero = {0.f, 0.f, 0.f, 0.f, 0.f, 0.f, 0.f, 0.f};
  v8f acc[2][4] = {{zero, zero, zero, zero}, {zero, zero, zero, zero}};

  const int nchunks = K >> 5;
  if (wv == 0) tdm_load_tile_128x32(lds0, wbase);   // prologue: chunk 0

  for (int i = 0; i < nchunks; ++i) {
    const int kc = i << 5;
    if (wv == 0) {
      if (i + 1 < nchunks) {
        // issue next chunk into the other buffer, then wait for current one
        tdm_load_tile_128x32((i & 1) ? lds0 : lds1,
                             wbase + (unsigned long long)(kc + 32) * 2u);
        __builtin_amdgcn_s_wait_tensorcnt(1);
      } else {
        __builtin_amdgcn_s_wait_tensorcnt(0);
      }
    }
    __syncthreads();  // current buffer ready for all waves

    v16bf a0 = frag_cat(*(const v8bf*)(arow0 + kc + half * 8),
                        *(const v8bf*)(arow0 + kc + 16 + half * 8));
    v16bf a1 = frag_cat(*(const v8bf*)(arow1 + kc + half * 8),
                        *(const v8bf*)(arow1 + kc + 16 + half * 8));
    const bf16_t* tb = &Wtile[i & 1][0][0];
#pragma unroll
    for (int g = 0; g < 4; ++g) {
      const bf16_t* wr = tb + (size_t)(nw + g * 16 + lr) * 32;
      v16bf bfv = frag_cat(*(const v8bf*)(wr + half * 8),
                           *(const v8bf*)(wr + 16 + half * 8));
      acc[0][g] = wmma_bf16(a0, bfv, acc[0][g]);
      acc[1][g] = wmma_bf16(a1, bfv, acc[1][g]);
    }
    __syncthreads();  // everyone done reading before buffer is overwritten
  }

#pragma unroll
  for (int sub = 0; sub < 2; ++sub) {
#pragma unroll
    for (int g = 0; g < 4; ++g) {
#pragma unroll
      for (int r = 0; r < 8; ++r) {
        C[(size_t)(m0 + sub * 16 + r + half * 8) * N +
          (n0blk + nw + g * 16 + lr)] = acc[sub][g][r];
      }
    }
  }
}

// ---------------------------------------------------------------------------
// RoPE + head split. Reads fp32 projections in [B,S,D_MODEL], writes:
//   Qh, Kh : bf16 [B,H,S,D_K]   (RoPE applied)
//   Vt     : bf16 [B,H,D_K,S]   (transposed for PV B-fragments)
// ---------------------------------------------------------------------------
__global__ void rope_reshape_kernel(const float* __restrict__ Qf,
                                    const float* __restrict__ Kf,
                                    const float* __restrict__ Vf,
                                    bf16_t* __restrict__ Qh,
                                    bf16_t* __restrict__ Kh,
                                    bf16_t* __restrict__ Vt) {
  int idx = blockIdx.x * blockDim.x + threadIdx.x;
  if (idx >= BATCH * SEQ * NUM_HEADS * 32) return;
  int k = idx & 31;
  int h = (idx >> 5) & (NUM_HEADS - 1);
  int s = (idx >> 9) & (SEQ - 1);
  int b = idx >> 20;

  size_t src = ((size_t)(b * SEQ + s)) * D_MODEL + h * D_K + 2 * k;
  float ang = (float)s * __powf(THETA, -(float)k / 32.0f);
  float sn, cs;
  __sincosf(ang, &sn, &cs);

  size_t dsth = (((size_t)(b * NUM_HEADS + h)) * SEQ + s) * D_K + 2 * k;
  float q0v = Qf[src], q1v = Qf[src + 1];
  Qh[dsth]     = (bf16_t)(q0v * cs - q1v * sn);
  Qh[dsth + 1] = (bf16_t)(q0v * sn + q1v * cs);
  float k0v = Kf[src], k1v = Kf[src + 1];
  Kh[dsth]     = (bf16_t)(k0v * cs - k1v * sn);
  Kh[dsth + 1] = (bf16_t)(k0v * sn + k1v * cs);

  size_t dv = (((size_t)(b * NUM_HEADS + h)) * D_K + 2 * k) * SEQ + s;
  Vt[dv]       = (bf16_t)Vf[src];
  Vt[dv + SEQ] = (bf16_t)Vf[src + 1];
}

// ---------------------------------------------------------------------------
// Flash attention (causal, online softmax). One wave per (b,h,16-row q tile).
// scoresT = K_tile(16x64) @ Q^T  => C layout: lane lr holds query q0+lr,
// rows are keys -> softmax stats are per-lane + one shfl_xor(16).
// P goes through LDS to become the A fragment of P(16x32) @ V(32x64).
// ---------------------------------------------------------------------------
__global__ void flash_attn_kernel(const bf16_t* __restrict__ Qh,
                                  const bf16_t* __restrict__ Kh,
                                  const bf16_t* __restrict__ Vt,
                                  bf16_t* __restrict__ Oh) {
  const int lane = threadIdx.x & 31;
  const int w    = (threadIdx.x >> 5) & 3;
  const int wid  = (int)((blockIdx.x * blockDim.x + threadIdx.x) >> 5);
  const int qt   = wid & (SEQ / 16 - 1);
  const int h    = (wid >> 7) & (NUM_HEADS - 1);
  const int b    = wid >> 11;
  const int q0   = qt << 4;
  const int half = lane >> 4;
  const int lr   = lane & 15;

  __shared__ __attribute__((aligned(16))) bf16_t Pb[4][16][40];

  const size_t bh = (size_t)(b * NUM_HEADS + h);
  const bf16_t* qbase = Qh + (bh * SEQ + q0) * D_K;
  const bf16_t* kbase = Kh + bh * SEQ * D_K;
  const bf16_t* vbase = Vt + bh * D_K * SEQ;

  v16bf qf[2];
#pragma unroll
  for (int dc = 0; dc < 2; ++dc) {
    qf[dc] = frag_cat(*(const v8bf*)(qbase + (size_t)lr * D_K + dc * 32 + half * 8),
                      *(const v8bf*)(qbase + (size_t)lr * D_K + dc * 32 + 16 + half * 8));
  }

  const v8f zero = {0.f, 0.f, 0.f, 0.f, 0.f, 0.f, 0.f, 0.f};
  v8f o[4] = {zero, zero, zero, zero};
  float l = 0.f, mprev = NEG_BIG;
  const float scale = 0.125f;  // 1/sqrt(64)

  for (int k0 = 0; k0 < q0 + 16; k0 += 32) {
    // prefetch next chunk's K rows (global_prefetch_b8)
    if (k0 + 32 < q0 + 16) {
      __builtin_prefetch(kbase + (size_t)(k0 + 32 + lane) * D_K, 0, 0);
    }
    float sv[2][8];
#pragma unroll
    for (int t = 0; t < 2; ++t) {
      v8f st = zero;
#pragma unroll
      for (int dc = 0; dc < 2; ++dc) {
        const bf16_t* kr = kbase + (size_t)(k0 + t * 16 + lr) * D_K + dc * 32;
        v16bf kf = frag_cat(*(const v8bf*)(kr + half * 8),
                            *(const v8bf*)(kr + 16 + half * 8));
        st = wmma_bf16(kf, qf[dc], st);
      }
#pragma unroll
      for (int r = 0; r < 8; ++r) {
        int key = k0 + t * 16 + r + 8 * half;
        float s = st[r] * scale;
        sv[t][r] = (key <= q0 + lr) ? s : NEG_BIG;
      }
    }
    float lmax = NEG_BIG;
#pragma unroll
    for (int t = 0; t < 2; ++t)
#pragma unroll
      for (int r = 0; r < 8; ++r) lmax = fmaxf(lmax, sv[t][r]);
    lmax = fmaxf(lmax, __shfl_xor(lmax, 16));
    float mnew = fmaxf(mprev, lmax);
    float sf = __expf(mprev - mnew);

    float psum = 0.f;
#pragma unroll
    for (int t = 0; t < 2; ++t) {
      v8bf pk;
#pragma unroll
      for (int r = 0; r < 8; ++r) {
        float p = __expf(sv[t][r] - mnew);
        psum += p;
        pk[r] = (bf16_t)p;
      }
      *(v8bf*)(&Pb[w][lr][t * 16 + half * 8]) = pk;  // P[q][key]
    }
    psum += __shfl_xor(psum, 16);
    l = l * sf + psum;

#pragma unroll
    for (int r = 0; r < 8; ++r) {
      float srow = __shfl(sf, r + 8 * half);
#pragma unroll
      for (int g = 0; g < 4; ++g) o[g][r] *= srow;
    }

    asm volatile("s_wait_dscnt 0" ::: "memory");

    v16bf pa = frag_cat(*(const v8bf*)(&Pb[w][lr][half * 8]),
                        *(const v8bf*)(&Pb[w][lr][16 + half * 8]));
#pragma unroll
    for (int g = 0; g < 4; ++g) {
      const bf16_t* vr = vbase + (size_t)(g * 16 + lr) * SEQ + k0;
      v16bf vf = frag_cat(*(const v8bf*)(vr + half * 8),
                          *(const v8bf*)(vr + 16 + half * 8));
      o[g] = wmma_bf16(pa, vf, o[g]);
    }
    mprev = mnew;
  }

  float inv = 1.0f / l;
#pragma unroll
  for (int r = 0; r < 8; ++r) {
    float rinv = __shfl(inv, r + 8 * half);
    int srow = q0 + r + 8 * half;
    bf16_t* orow = Oh + ((size_t)(b * SEQ + srow)) * D_MODEL + h * D_K;
#pragma unroll
    for (int g = 0; g < 4; ++g) orow[g * 16 + lr] = (bf16_t)(o[g][r] * rinv);
  }
}

// ---------------------------------------------------------------------------
extern "C" void kernel_launch(void* const* d_in, const int* in_sizes, int n_in,
                              void* d_out, int out_size, void* d_ws,
                              size_t ws_size, hipStream_t stream) {
  (void)in_sizes; (void)n_in; (void)out_size; (void)ws_size;
  const float* x  = (const float*)d_in[0];
  const float* Wq = (const float*)d_in[1];
  const float* Wk = (const float*)d_in[2];
  const float* Wv = (const float*)d_in[3];
  const float* Wo = (const float*)d_in[4];

  char* ws = (char*)d_ws;
  const size_t MiB = 1024 * 1024;
  bf16_t* xh  = (bf16_t*)(ws + 0);
  bf16_t* Wqh = (bf16_t*)(ws + 8  * MiB);
  bf16_t* Wkh = (bf16_t*)(ws + 10 * MiB);
  bf16_t* Wvh = (bf16_t*)(ws + 12 * MiB);
  bf16_t* Woh = (bf16_t*)(ws + 14 * MiB);
  float*  Qf  = (float*) (ws + 16 * MiB);
  float*  Kf  = (float*) (ws + 32 * MiB);
  float*  Vf  = (float*) (ws + 48 * MiB);
  bf16_t* Qhp = (bf16_t*)(ws + 64 * MiB);
  bf16_t* Khp = (bf16_t*)(ws + 72 * MiB);
  bf16_t* Vtp = (bf16_t*)(ws + 80 * MiB);
  bf16_t* Ohp = (bf16_t*)(ws + 88 * MiB);

  const int M = BATCH * SEQ;   // 4096
  const int N = D_MODEL;       // 1024
  const int K = D_MODEL;       // 1024

  f32_to_bf16_kernel<<<(M * K + 255) / 256, 256, 0, stream>>>(x, xh, M * K);
  f32_to_bf16_kernel<<<(N * K + 255) / 256, 256, 0, stream>>>(Wq, Wqh, N * K);
  f32_to_bf16_kernel<<<(N * K + 255) / 256, 256, 0, stream>>>(Wk, Wkh, N * K);
  f32_to_bf16_kernel<<<(N * K + 255) / 256, 256, 0, stream>>>(Wv, Wvh, N * K);
  f32_to_bf16_kernel<<<(N * K + 255) / 256, 256, 0, stream>>>(Wo, Woh, N * K);

  // block computes 64x128 -> (M/64)*(N/128) = 512 blocks of 128 threads
  const int gemm_blocks = (M / 64) * (N / 128);
  gemm_xwt_kernel<<<gemm_blocks, 128, 0, stream>>>(xh, Wqh, Qf, M, N, K);
  gemm_xwt_kernel<<<gemm_blocks, 128, 0, stream>>>(xh, Wkh, Kf, M, N, K);
  gemm_xwt_kernel<<<gemm_blocks, 128, 0, stream>>>(xh, Wvh, Vf, M, N, K);

  const int rope_threads = BATCH * SEQ * NUM_HEADS * 32;
  rope_reshape_kernel<<<(rope_threads + 255) / 256, 256, 0, stream>>>(
      Qf, Kf, Vf, Qhp, Khp, Vtp);

  const int fa_blocks = BATCH * NUM_HEADS * (SEQ / 16) / 4;
  flash_attn_kernel<<<fa_blocks, 128, 0, stream>>>(Qhp, Khp, Vtp, Ohp);

  gemm_xwt_kernel<<<gemm_blocks, 128, 0, stream>>>(Ohp, Woh, (float*)d_out,
                                                   M, N, K);
}
